// NyAttenLayer_75737453298226
// MI455X (gfx1250) — matrix operand order repaired and checked by reading, not compiled, not run on hardware
//
#include <hip/hip_runtime.h>
#include <hip/hip_bf16.h>

// ---------------------------------------------------------------------------
// Types for CDNA5 WMMA / TDM
// ---------------------------------------------------------------------------
typedef __bf16   v16bf __attribute__((ext_vector_type(16)));
typedef float    v8f   __attribute__((ext_vector_type(8)));
typedef float    v2f   __attribute__((ext_vector_type(2)));
typedef unsigned v4u   __attribute__((ext_vector_type(4)));
typedef int      v4i   __attribute__((ext_vector_type(4)));
typedef int      v8i   __attribute__((ext_vector_type(8)));

union FragBF { v16bf v; unsigned u[8]; };

static __device__ inline v8f wmma_bf16f32(v16bf a, v16bf b, v8f c) {
  // v_wmma_f32_16x16x32_bf16
  return __builtin_amdgcn_wmma_f32_16x16x32_bf16(false, a, false, b, (short)0, c, false, false);
}

static __device__ inline v8f wmma_f32x4(v2f a, v2f b, v8f c) {
#if defined(__has_builtin) && __has_builtin(__builtin_amdgcn_wmma_f32_16x16x4_f32)
  // v_wmma_f32_16x16x4_f32
  return __builtin_amdgcn_wmma_f32_16x16x4_f32(false, a, false, b, (short)0, c, false, false);
#else
  c[0] += a[0] * b[0];
  c[1] += a[1] * b[1];
  return c;
#endif
}

// ---------------------------------------------------------------------------
// Tensor Data Mover: 2-D tile global -> LDS (CDNA5 ISA sect 8).
// D# group0: count=1 | lds_addr | global_addr[56:0] | type=2.
// D# group1: data_size=1 (2B) + optional LDS padding (pad_interval code 3 =
// every 16 DWORDs, pad_amount code 3 = 4 DWORDs -> turns a 32-elem bf16 row
// into the [rows][40] bank-padded LDS layout used by the fragment loaders).
// tensor dims == tile dims (no OOB), tensor_dim0_stride = global row stride.
// Issued by one wave (TDM ignores EXEC); completion via s_wait_tensorcnt.
// ---------------------------------------------------------------------------
static __device__ inline unsigned lds_addr_of(const void* p) {
  // LDS aperture: flat addr bits [31:0] are the LDS byte offset (ISA 10.2).
  return (unsigned)(unsigned long long)p;
}

static __device__ inline void tdm_load_2d_bf16(unsigned lds_off, const void* g,
                                               unsigned tile_w, unsigned tile_h,
                                               unsigned stride_elems, bool pad16dw4) {
  const unsigned long long ga = (unsigned long long)g;
  v4u g0;
  g0[0] = 1u;                                               // count=1, user D#
  g0[1] = lds_off;                                          // lds_addr (bytes)
  g0[2] = (unsigned)ga;                                     // global_addr[31:0]
  g0[3] = (unsigned)((ga >> 32) & 0x01FFFFFFull) | (2u << 30);  // ga[56:32] | type=2
  unsigned w0 = (1u << 16);                                 // data_size=1 (2 bytes)
  if (pad16dw4) w0 |= (1u << 20) | (3u << 22) | (3u << 25); // pad_en | every 16DW | +4DW
  v8i g1;
  g1[0] = (int)w0;
  g1[1] = (int)(tile_w << 16);                              // tensor_dim0 lo16
  g1[2] = (int)((tile_w >> 16) | (tile_h << 16));           // tensor_dim0 hi | dim1 lo
  g1[3] = (int)((tile_h >> 16) | (tile_w << 16));           // dim1 hi | tile_dim0
  g1[4] = (int)(tile_h & 0xffff);                           // tile_dim1 (tile_dim2=0)
  g1[5] = (int)stride_elems;                                // tensor_dim0_stride lo32
  g1[6] = 0;
  g1[7] = 0;
  const v4i z4 = {0, 0, 0, 0};
#if __clang_major__ >= 23
  const v8i z8v = {0, 0, 0, 0, 0, 0, 0, 0};
  __builtin_amdgcn_tensor_load_to_lds(g0, g1, z4, z4, z8v, 0);
#else
  __builtin_amdgcn_tensor_load_to_lds(g0, g1, z4, z4, 0);
#endif
}

// ---------------------------------------------------------------------------
// Fragment loaders per ISA 7.12.2 layouts (wave32).
// ---------------------------------------------------------------------------
static __device__ inline v16bf load_fragA_bf16(const __bf16* base, int stride, int lane) {
  FragBF f;
  const int m = lane & 15, hi = lane >> 4;
  const __bf16* row = base + m * stride;
#pragma unroll
  for (int e = 0; e < 8; ++e) {
    const int kb = (e < 4) ? (hi * 8 + 2 * e) : (16 + hi * 8 + 2 * (e - 4));
    f.u[e] = *(const unsigned*)(row + kb);
  }
  return f.v;
}

static __device__ inline v16bf load_fragB_bf16(const __bf16* base, int stride, int lane, int kofs) {
  FragBF f;
  const int n = lane & 15, hi = lane >> 4;
  const __bf16* row = base + n * stride + kofs + hi * 8;
#pragma unroll
  for (int e = 0; e < 8; ++e) f.u[e] = *(const unsigned*)(row + ((e < 4) ? 2 * e : 16 + 2 * (e - 4)));
  return f.v;
}

static __device__ inline v2f load_frag_f32(const float* base, int stride, int lane, int kofs) {
  const int m = lane & 15, hi = lane >> 4;
  const float* p = base + m * stride + kofs + 2 * hi;
  v2f r;
  r[0] = p[0];
  r[1] = p[1];
  return r;
}

static __device__ inline float hmax16(float v) {
  v = fmaxf(v, __shfl_xor(v, 1, 32));
  v = fmaxf(v, __shfl_xor(v, 2, 32));
  v = fmaxf(v, __shfl_xor(v, 4, 32));
  v = fmaxf(v, __shfl_xor(v, 8, 32));
  return v;
}
static __device__ inline float hsum16(float v) {
  v += __shfl_xor(v, 1, 32);
  v += __shfl_xor(v, 2, 32);
  v += __shfl_xor(v, 4, 32);
  v += __shfl_xor(v, 8, 32);
  return v;
}

// ---------------------------------------------------------------------------
// Problem constants
// ---------------------------------------------------------------------------
#define BSZ   4
#define HEADS 8
#define SEQ   8192
#define DHEAD 64
#define DIMF  512
#define MLAND 256
#define BH    32
#define ROWS  32768

__global__ void k_f32_to_bf16(const float* __restrict__ x, __bf16* __restrict__ y) {
  const long i = (long)blockIdx.x * 256 + threadIdx.x;
  y[i] = (__bf16)x[i];
}

// ---------------------------------------------------------------------------
// Generic bf16 GEMM (block 128x256, wave 64x64, BK=32).  A tile staged via
// TDM with LDS padding; B tile via TDM when BT (already [N,K]) or manual
// transpose staging otherwise.
// EPI: 0 bf16 store | 1 qkv scatter | 2 f32 +bias+resid | 3 f32 (batched)
// ---------------------------------------------------------------------------
template <int EPI, bool BT>
__global__ __launch_bounds__(256) void k_gemm_bf16(
    const __bf16* __restrict__ A, const __bf16* __restrict__ Bm,
    int lda, int ldb, int K, long sA, long sB, long sC, float alpha,
    float* __restrict__ outF, __bf16* __restrict__ outB, int ldc,
    const float* __restrict__ bias, const float* __restrict__ resid,
    __bf16* __restrict__ q_out, __bf16* __restrict__ k_out, __bf16* __restrict__ v_out) {
  __shared__ __attribute__((aligned(16))) __bf16 As[128][40];
  __shared__ __attribute__((aligned(16))) __bf16 Bs[256][40];

  const int t = threadIdx.x, lane = t & 31, wid = t >> 5;
  const int wr = wid >> 2, wc = wid & 3;
  const int rb = blockIdx.x * 128, cb = blockIdx.y * 256;
  const int zb = blockIdx.z;
  A += (long)zb * sA;
  Bm += (long)zb * sB;

  v8f acc[4][4];
  const v8f z8 = {0.f, 0.f, 0.f, 0.f, 0.f, 0.f, 0.f, 0.f};
#pragma unroll
  for (int i = 0; i < 4; ++i)
#pragma unroll
    for (int j = 0; j < 4; ++j) acc[i][j] = z8;

  for (int k0 = 0; k0 < K; k0 += 32) {
    __syncthreads();
    if (!BT) {
#pragma unroll
      for (int ch = t; ch < 1024; ch += 256) {
        const int kk = ch >> 5, ng = ch & 31;
        const uint4 dv = *(const uint4*)(Bm + (long)(k0 + kk) * ldb + cb + ng * 8);
        const __bf16* p = (const __bf16*)&dv;
#pragma unroll
        for (int i = 0; i < 8; ++i) Bs[ng * 8 + i][kk] = p[i];
      }
    }
    if (t < 32) {  // wave 0 drives the Tensor Data Mover
      tdm_load_2d_bf16(lds_addr_of(&As[0][0]), A + (long)rb * lda + k0, 32, 128, lda, true);
      if (BT)
        tdm_load_2d_bf16(lds_addr_of(&Bs[0][0]), Bm + (long)cb * ldb + k0, 32, 256, ldb, true);
      __builtin_amdgcn_s_wait_tensorcnt(0);
    }
    if (k0 + 32 < K)  // L2 hint for the next A tile (global_prefetch_b8)
      __builtin_prefetch(A + (long)(rb + (t >> 1)) * lda + k0 + 32, 0, 1);
    __syncthreads();

    v16bf af[4], bfv[4];
#pragma unroll
    for (int i = 0; i < 4; ++i) af[i] = load_fragA_bf16(&As[wr * 64 + i * 16][0], 40, lane);
#pragma unroll
    for (int j = 0; j < 4; ++j) bfv[j] = load_fragB_bf16(&Bs[wc * 64 + j * 16][0], 40, lane, 0);
#pragma unroll
    for (int i = 0; i < 4; ++i)
#pragma unroll
      for (int j = 0; j < 4; ++j) acc[i][j] = wmma_bf16f32(af[i], bfv[j], acc[i][j]);
  }

  const int hi = lane >> 4, ln = lane & 15;
#pragma unroll
  for (int i = 0; i < 4; ++i)
#pragma unroll
    for (int j = 0; j < 4; ++j)
#pragma unroll
      for (int r = 0; r < 8; ++r) {
        const int row = rb + wr * 64 + i * 16 + r + 8 * hi;
        const int col = cb + wc * 64 + j * 16 + ln;
        float v = acc[i][j][r] * alpha;
        if constexpr (EPI == 0) {
          outB[(long)row * ldc + col] = (__bf16)v;
        } else if constexpr (EPI == 3) {
          (outF + (long)zb * sC)[(long)row * ldc + col] = v;
        } else if constexpr (EPI == 2) {
          const long idx = (long)row * ldc + col;
          outF[idx] = v + bias[col] + resid[idx];
        } else {  // qkv scatter
          const int part = col >> 9;
          const int inner = col & 511;
          const int h = inner >> 6, d = inner & 63;
          const int b = row >> 13, n = row & 8191;
          const long idx = (((long)(b * HEADS + h)) * SEQ + n) * DHEAD + d;
          __bf16* dst = (part == 0) ? q_out : (part == 1) ? k_out : v_out;
          if (part == 0) v *= 0.125f;  // d_head^-0.5
          dst[idx] = (__bf16)v;
        }
      }
}

// ---------------------------------------------------------------------------
// Landmark mean-pooling
// ---------------------------------------------------------------------------
__global__ void k_pool(const __bf16* __restrict__ q, const __bf16* __restrict__ k,
                       __bf16* __restrict__ ql, __bf16* __restrict__ kl) {
  const int bhm = blockIdx.x;
  const int bh = bhm >> 8, mi = bhm & 255;
  const int d = threadIdx.x;
  const long base = ((long)bh * SEQ + mi * 32) * DHEAD + d;
  float aq = 0.f, ak = 0.f;
#pragma unroll
  for (int i = 0; i < 32; ++i) {
    aq += (float)q[base + (long)i * DHEAD];
    ak += (float)k[base + (long)i * DHEAD];
  }
  const long o = ((long)bh * MLAND + mi) * DHEAD + d;
  ql[o] = (__bf16)(aq * (1.f / 32.f));
  kl[o] = (__bf16)(ak * (1.f / 32.f));
}

// ---------------------------------------------------------------------------
// Row softmax width 256 (attn2), in place.
// ---------------------------------------------------------------------------
__global__ __launch_bounds__(256) void k_softmax256(float* __restrict__ a) {
  __shared__ float redm[8], reds[8];
  const long r = blockIdx.x;
  const int t = threadIdx.x, lane = t & 31, w = t >> 5;
  float* row = a + r * 256;
  const float v = row[t];
  float m = v;
  for (int msk = 1; msk < 32; msk <<= 1) m = fmaxf(m, __shfl_xor(m, msk, 32));
  if (lane == 0) redm[w] = m;
  __syncthreads();
  m = redm[0];
#pragma unroll
  for (int i = 1; i < 8; ++i) m = fmaxf(m, redm[i]);
  const float e = __expf(v - m);
  float s = e;
  for (int msk = 1; msk < 32; msk <<= 1) s += __shfl_xor(s, msk, 32);
  if (lane == 0) reds[w] = s;
  __syncthreads();
  s = 0.f;
#pragma unroll
  for (int i = 0; i < 8; ++i) s += reds[i];
  row[t] = e / s;
}

__global__ __launch_bounds__(256) void k_redmax(const float* __restrict__ a2, float* __restrict__ scal) {
  __shared__ float red[8];
  const int x = blockIdx.x, which = blockIdx.y;
  const int bh = x >> 8, i = x & 255;
  const int t = threadIdx.x, lane = t & 31, w = t >> 5;
  const float* m = a2 + (long)bh * 65536;
  float v = fabsf(which == 0 ? m[i * 256 + t] : m[t * 256 + i]);
  for (int msk = 1; msk < 32; msk <<= 1) v += __shfl_xor(v, msk, 32);
  if (lane == 0) red[w] = v;
  __syncthreads();
  if (t == 0) {
    float s = 0.f;
#pragma unroll
    for (int j = 0; j < 8; ++j) s += red[j];
    atomicMax((unsigned int*)(scal + which), __float_as_uint(s));
  }
}

__global__ void k_pinv_init(const float* __restrict__ a2, const float* __restrict__ scal,
                            float* __restrict__ z) {
  const long idx = (long)blockIdx.x * 256 + threadIdx.x;
  const int bh = idx >> 16, rem = idx & 65535;
  const int i = rem >> 8, j = rem & 255;
  const float inv = 1.f / (scal[0] * scal[1]);
  z[idx] = a2[(long)bh * 65536 + j * 256 + i] * inv;
}

__global__ void k_diag_sub(const float* __restrict__ s, float* __restrict__ d, float c) {
  const long idx = (long)blockIdx.x * 256 + threadIdx.x;
  const int rem = idx & 65535;
  const int i = rem >> 8, j = rem & 255;
  d[idx] = ((i == j) ? c : 0.f) - s[idx];
}

__global__ void k_zT_bf16(const float* __restrict__ z, __bf16* __restrict__ zT) {
  const long idx = (long)blockIdx.x * 256 + threadIdx.x;
  const int bh = idx >> 16, rem = idx & 65535;
  const int n = rem >> 8, k = rem & 255;
  zT[idx] = (__bf16)z[(long)bh * 65536 + k * 256 + n];
}

// ---------------------------------------------------------------------------
// Batched f32 WMMA GEMM 256x256x256 (pinv Newton-Schulz iterations).
// ---------------------------------------------------------------------------
__global__ __launch_bounds__(256) void k_pinv_gemm(const float* __restrict__ A,
                                                   const float* __restrict__ B,
                                                   float* __restrict__ C, float alpha) {
  __shared__ __attribute__((aligned(16))) float Asf[128][36];
  __shared__ __attribute__((aligned(16))) float BsT[256][36];
  const int t = threadIdx.x, lane = t & 31, wid = t >> 5;
  const int wr = wid >> 2, wc = wid & 3;
  const long zo = (long)blockIdx.z * 65536;
  A += zo;
  B += zo;
  C += zo;
  const int rb = blockIdx.x * 128;

  v8f acc[4][4];
  const v8f z8 = {0.f, 0.f, 0.f, 0.f, 0.f, 0.f, 0.f, 0.f};
#pragma unroll
  for (int i = 0; i < 4; ++i)
#pragma unroll
    for (int j = 0; j < 4; ++j) acc[i][j] = z8;

  for (int k0 = 0; k0 < 256; k0 += 32) {
    __syncthreads();
#pragma unroll
    for (int ch = t; ch < 1024; ch += 256) {
      const int row = ch >> 3, cg = ch & 7;
      *(float4*)&Asf[row][cg * 4] = *(const float4*)(A + (long)(rb + row) * 256 + k0 + cg * 4);
    }
#pragma unroll
    for (int ch = t; ch < 2048; ch += 256) {
      const int kk = ch >> 6, ng = ch & 63;
      const float4 dv = *(const float4*)(B + (long)(k0 + kk) * 256 + ng * 4);
      const float* p = (const float*)&dv;
#pragma unroll
      for (int i = 0; i < 4; ++i) BsT[ng * 4 + i][kk] = p[i];
    }
    __syncthreads();
#pragma unroll
    for (int kk = 0; kk < 8; ++kk) {
      v2f af[4], bfv[4];
#pragma unroll
      for (int i = 0; i < 4; ++i) af[i] = load_frag_f32(&Asf[wr * 64 + i * 16][0], 36, lane, kk * 4);
#pragma unroll
      for (int j = 0; j < 4; ++j) bfv[j] = load_frag_f32(&BsT[wc * 64 + j * 16][0], 36, lane, kk * 4);
#pragma unroll
      for (int i = 0; i < 4; ++i)
#pragma unroll
        for (int j = 0; j < 4; ++j) acc[i][j] = wmma_f32x4(af[i], bfv[j], acc[i][j]);
    }
  }
  const int hi = lane >> 4, ln = lane & 15;
#pragma unroll
  for (int i = 0; i < 4; ++i)
#pragma unroll
    for (int j = 0; j < 4; ++j)
#pragma unroll
      for (int r = 0; r < 8; ++r)
        C[(long)(rb + wr * 64 + i * 16 + r + 8 * hi) * 256 + wc * 64 + j * 16 + ln] =
            alpha * acc[i][j][r];
}

// ---------------------------------------------------------------------------
// attn3v = softmax_n(q_l @ k^T) @ v, flash-style streaming softmax.
// ---------------------------------------------------------------------------
__global__ __launch_bounds__(128) void k_attn3v(const __bf16* __restrict__ ql,
                                                const __bf16* __restrict__ k,
                                                const __bf16* __restrict__ v,
                                                __bf16* __restrict__ a3vT) {
  __shared__ __attribute__((aligned(16))) __bf16 vT[64][264];
  __shared__ __attribute__((aligned(16))) __bf16 Pst[4][16][264];
  const int t = threadIdx.x, lane = t & 31, w = t >> 5;
  const int bh = blockIdx.y;
  const int qt = blockIdx.x * 4 + w;
  const int hi = lane >> 4, ln = lane & 15;
  const __bf16* kb = k + (long)bh * SEQ * DHEAD;
  const __bf16* vb = v + (long)bh * SEQ * DHEAD;
  const __bf16* qlrow = ql + ((long)bh * MLAND + qt * 16) * DHEAD;

  v16bf aql[2];
  aql[0] = load_fragA_bf16(qlrow, DHEAD, lane);
  aql[1] = load_fragA_bf16(qlrow + 32, DHEAD, lane);

  float mrun[8], lrun[8], scale[8];
#pragma unroll
  for (int r = 0; r < 8; ++r) { mrun[r] = -1e30f; lrun[r] = 0.f; }
  v8f oc[4];
  const v8f z8 = {0.f, 0.f, 0.f, 0.f, 0.f, 0.f, 0.f, 0.f};
#pragma unroll
  for (int j = 0; j < 4; ++j) oc[j] = z8;

  for (int c = 0; c < SEQ / 256; ++c) {
    const int nb = c * 256;
    __syncthreads();
#pragma unroll
    for (int ch = t; ch < 2048; ch += 128) {
      const int n = ch >> 3, dg = ch & 7;
      const uint4 dv = *(const uint4*)(vb + (long)(nb + n) * DHEAD + dg * 8);
      const __bf16* p = (const __bf16*)&dv;
#pragma unroll
      for (int i = 0; i < 8; ++i) vT[dg * 8 + i][n] = p[i];
    }
    __syncthreads();

    v8f sc[16];
#pragma unroll
    for (int ct = 0; ct < 16; ++ct) sc[ct] = z8;
#pragma unroll
    for (int ks = 0; ks < 2; ++ks)
#pragma unroll
      for (int ct = 0; ct < 16; ++ct) {
        const v16bf b = load_fragB_bf16(kb + (long)(nb + ct * 16) * DHEAD + ks * 32, DHEAD, lane, 0);
        sc[ct] = wmma_bf16f32(aql[ks], b, sc[ct]);
      }

#pragma unroll
    for (int r = 0; r < 8; ++r) {
      float mx = -1e30f;
#pragma unroll
      for (int ct = 0; ct < 16; ++ct) mx = fmaxf(mx, sc[ct][r]);
      mx = hmax16(mx);
      const float nm = fmaxf(mrun[r], mx);
      scale[r] = __expf(mrun[r] - nm);
      mrun[r] = nm;
      float s = 0.f;
#pragma unroll
      for (int ct = 0; ct < 16; ++ct) {
        const float p = __expf(sc[ct][r] - nm);
        s += p;
        Pst[w][r + 8 * hi][ct * 16 + ln] = (__bf16)p;
      }
      lrun[r] = lrun[r] * scale[r] + hsum16(s);
    }
#pragma unroll
    for (int j = 0; j < 4; ++j)
#pragma unroll
      for (int r = 0; r < 8; ++r) oc[j][r] *= scale[r];
    __syncthreads();

#pragma unroll
    for (int ks = 0; ks < 8; ++ks) {
      const v16bf a = load_fragA_bf16(&Pst[w][0][0] + ks * 32, 264, lane);
#pragma unroll
      for (int j = 0; j < 4; ++j) {
        const v16bf b = load_fragB_bf16(&vT[j * 16][0] + ks * 32, 264, lane, 0);
        oc[j] = wmma_bf16f32(a, b, oc[j]);
      }
    }
  }
  __bf16* outb = a3vT + (long)bh * DHEAD * MLAND;
#pragma unroll
  for (int j = 0; j < 4; ++j)
#pragma unroll
    for (int r = 0; r < 8; ++r) {
      const int d = j * 16 + ln, row = r + 8 * hi;
      outb[(long)d * MLAND + qt * 16 + row] = (__bf16)(oc[j][r] / lrun[r]);
    }
}

// ---------------------------------------------------------------------------
// Depthwise conv residual over sequence, per head: writes oh[b][n][h*64+d].
// ---------------------------------------------------------------------------
__global__ void k_conv(const __bf16* __restrict__ v, const float* __restrict__ w,
                       __bf16* __restrict__ oh) {
  const long idx = (long)blockIdx.x * 256 + threadIdx.x;
  const int d = idx & 63;
  const long r = idx >> 6;
  const int n = r & 8191;
  const int bh = r >> 13, b = bh >> 3, h = bh & 7;
  const __bf16* vb = v + (long)bh * SEQ * DHEAD + d;
  float acc = 0.f;
#pragma unroll
  for (int j = 0; j < 33; ++j) {
    const int nn = n + j - 16;
    if (0 <= nn && nn < SEQ) acc += w[h * 33 + j] * (float)vb[(long)nn * DHEAD];
  }
  oh[((long)b * SEQ + n) * DIMF + h * 64 + d] = (__bf16)acc;
}

// ---------------------------------------------------------------------------
// Fused attn1 chain.  Z+^T (128KB) and attn3v^T (32KB) staged once per block
// via two TDM descriptors (320KB WGP LDS); P/O1 round-trip LDS to reshape
// C-frag -> A-frag.
// ---------------------------------------------------------------------------
__global__ __launch_bounds__(256) void k_attn_fused(const __bf16* __restrict__ q,
                                                    const __bf16* __restrict__ kl,
                                                    const __bf16* __restrict__ zT,
                                                    const __bf16* __restrict__ a3vT,
                                                    __bf16* __restrict__ oh) {
  extern __shared__ __attribute__((aligned(16))) __bf16 smem[];
  __bf16* zS = smem;            // [256][256]
  __bf16* aS = zS + 256 * 256;  // [64][256]
  __bf16* pS = aS + 64 * 256;   // [8][16][264]
  const int t = threadIdx.x, lane = t & 31, w = t >> 5;
  const int bh = blockIdx.y, b = bh >> 3, h = bh & 7;
  const int nbase = blockIdx.x * 128 + w * 16;
  const int hi = lane >> 4, ln = lane & 15;

  if (t < 32) {  // wave 0 drives the Tensor Data Mover
    tdm_load_2d_bf16(lds_addr_of(zS), zT + (long)bh * 65536, 256, 256, 256, false);
    tdm_load_2d_bf16(lds_addr_of(aS), a3vT + (long)bh * 16384, 256, 64, 256, false);
    __builtin_amdgcn_s_wait_tensorcnt(0);
  }
  __syncthreads();

  const __bf16* qrow = q + ((long)bh * SEQ + nbase) * DHEAD;
  const __bf16* klb = kl + (long)bh * MLAND * DHEAD;
  const v8f z8 = {0.f, 0.f, 0.f, 0.f, 0.f, 0.f, 0.f, 0.f};

  // S = Q @ KL^T    [16 x 256]
  v8f sc[16];
#pragma unroll
  for (int ct = 0; ct < 16; ++ct) sc[ct] = z8;
#pragma unroll
  for (int ks = 0; ks < 2; ++ks) {
    const v16bf a = load_fragA_bf16(qrow + ks * 32, DHEAD, lane);
#pragma unroll
    for (int ct = 0; ct < 16; ++ct) {
      const v16bf bf = load_fragB_bf16(klb + (long)(ct * 16) * DHEAD + ks * 32, DHEAD, lane, 0);
      sc[ct] = wmma_bf16f32(a, bf, sc[ct]);
    }
  }

  // softmax rows -> P (bf16) in this wave's LDS slab
  __bf16* P = pS + (long)w * 16 * 264;
#pragma unroll
  for (int r = 0; r < 8; ++r) {
    float mx = -1e30f;
#pragma unroll
    for (int ct = 0; ct < 16; ++ct) mx = fmaxf(mx, sc[ct][r]);
    mx = hmax16(mx);
    float s = 0.f;
#pragma unroll
    for (int ct = 0; ct < 16; ++ct) {
      const float p = __expf(sc[ct][r] - mx);
      sc[ct][r] = p;
      s += p;
    }
    s = hsum16(s);
    const float inv = 1.f / s;
#pragma unroll
    for (int ct = 0; ct < 16; ++ct)
      P[(r + 8 * hi) * 264 + ct * 16 + ln] = (__bf16)(sc[ct][r] * inv);
  }

  // O1 = P @ Z+   [16 x 256]
  v8f o1[16];
#pragma unroll
  for (int ct = 0; ct < 16; ++ct) o1[ct] = z8;
#pragma unroll
  for (int ks = 0; ks < 8; ++ks) {
    const v16bf a = load_fragA_bf16(P + ks * 32, 264, lane);
#pragma unroll
    for (int ct = 0; ct < 16; ++ct) {
      const v16bf bf = load_fragB_bf16(zS + (long)(ct * 16) * 256 + ks * 32, 256, lane, 0);
      o1[ct] = wmma_bf16f32(a, bf, o1[ct]);
    }
  }
  // O1 -> bf16 -> LDS (reuse P slab; same-wave DS ops stay in order)
#pragma unroll
  for (int ct = 0; ct < 16; ++ct)
#pragma unroll
    for (int r = 0; r < 8; ++r)
      P[(r + 8 * hi) * 264 + ct * 16 + ln] = (__bf16)o1[ct][r];

  // O = O1 @ attn3v   [16 x 64]
  v8f oc[4];
#pragma unroll
  for (int j = 0; j < 4; ++j) oc[j] = z8;
#pragma unroll
  for (int ks = 0; ks < 8; ++ks) {
    const v16bf a = load_fragA_bf16(P + ks * 32, 264, lane);
#pragma unroll
    for (int j = 0; j < 4; ++j) {
      const v16bf bf = load_fragB_bf16(aS + (long)(j * 16) * 256 + ks * 32, 256, lane, 0);
      oc[j] = wmma_bf16f32(a, bf, oc[j]);
    }
  }
  // add onto conv residual already in oh
#pragma unroll
  for (int j = 0; j < 4; ++j)
#pragma unroll
    for (int r = 0; r < 8; ++r) {
      const int n = nbase + r + 8 * hi;
      const int d = j * 16 + ln;
      const long idx = ((long)b * SEQ + n) * DIMF + h * 64 + d;
      oh[idx] = (__bf16)((float)oh[idx] + oc[j][r]);
    }
}

// ---------------------------------------------------------------------------
// Host orchestration
// ---------------------------------------------------------------------------
extern "C" void kernel_launch(void* const* d_in, const int* in_sizes, int n_in,
                              void* d_out, int out_size, void* d_ws, size_t ws_size,
                              hipStream_t stream) {
  const float* x      = (const float*)d_in[0];
  const float* w_qkv  = (const float*)d_in[1];
  const float* w_out  = (const float*)d_in[2];
  const float* b_out  = (const float*)d_in[3];
  const float* conv_w = (const float*)d_in[4];
  float* out = (float*)d_out;

  char* ws = (char*)d_ws;
  size_t off = 0;
  auto alloc = [&](size_t bytes) {
    void* p = ws + off;
    off = (off + bytes + 255) & ~(size_t)255;
    return p;
  };
  float*  scal  = (float*)alloc(256);
  __bf16* xb    = (__bf16*)alloc((size_t)ROWS * DIMF * 2);
  __bf16* wqkvb = (__bf16*)alloc((size_t)DIMF * 1536 * 2);
  __bf16* woutb = (__bf16*)alloc((size_t)DIMF * DIMF * 2);
  __bf16* qb    = (__bf16*)alloc((size_t)BH * SEQ * DHEAD * 2);
  __bf16* kb    = (__bf16*)alloc((size_t)BH * SEQ * DHEAD * 2);
  __bf16* vb    = (__bf16*)alloc((size_t)BH * SEQ * DHEAD * 2);
  __bf16* qlb   = (__bf16*)alloc((size_t)BH * MLAND * DHEAD * 2);
  __bf16* klb   = (__bf16*)alloc((size_t)BH * MLAND * DHEAD * 2);
  float*  attn2 = (float*)alloc((size_t)BH * MLAND * MLAND * 4);
  float*  zb0   = (float*)alloc((size_t)BH * MLAND * MLAND * 4);
  float*  zb1   = (float*)alloc((size_t)BH * MLAND * MLAND * 4);
  float*  tb2   = (float*)alloc((size_t)BH * MLAND * MLAND * 4);
  float*  tb3   = (float*)alloc((size_t)BH * MLAND * MLAND * 4);
  float*  tb4   = (float*)alloc((size_t)BH * MLAND * MLAND * 4);
  __bf16* zTb   = (__bf16*)alloc((size_t)BH * MLAND * MLAND * 2);
  __bf16* a3vT  = (__bf16*)alloc((size_t)BH * DHEAD * MLAND * 2);
  __bf16* oh    = (__bf16*)alloc((size_t)ROWS * DIMF * 2);

  k_f32_to_bf16<<<(ROWS * DIMF) / 256, 256, 0, stream>>>(x, xb);
  k_f32_to_bf16<<<(DIMF * 1536) / 256, 256, 0, stream>>>(w_qkv, wqkvb);
  k_f32_to_bf16<<<(DIMF * DIMF) / 256, 256, 0, stream>>>(w_out, woutb);

  // qkv projection, scatter into per-head layout (q scaled by d^-0.5)
  k_gemm_bf16<1, false><<<dim3(ROWS / 128, 1536 / 256, 1), 256, 0, stream>>>(
      xb, wqkvb, DIMF, 1536, DIMF, 0, 0, 0, 1.f,
      nullptr, nullptr, 0, nullptr, nullptr, qb, kb, vb);

  k_pool<<<BH * MLAND, DHEAD, 0, stream>>>(qb, kb, qlb, klb);

  // attn2 scores (batched, B transposed) + softmax
  k_gemm_bf16<3, true><<<dim3(MLAND / 128, 1, BH), 256, 0, stream>>>(
      qlb, klb, DHEAD, DHEAD, DHEAD,
      (long)MLAND * DHEAD, (long)MLAND * DHEAD, (long)MLAND * MLAND, 1.f,
      attn2, nullptr, MLAND, nullptr, nullptr, nullptr, nullptr, nullptr);
  k_softmax256<<<BH * MLAND, 256, 0, stream>>>(attn2);

  hipMemsetAsync(scal, 0, 8, stream);
  k_redmax<<<dim3(BH * MLAND, 2), 256, 0, stream>>>(attn2, scal);
  k_pinv_init<<<(BH * MLAND * MLAND) / 256, 256, 0, stream>>>(attn2, scal, zb0);

  // Newton-Schulz: z = 0.25 * z @ (13I - xz @ (15I - xz @ (7I - xz)))
  float* zin = zb0;
  float* zout = zb1;
  const dim3 pg(2, 1, BH);
  const int eb = (BH * MLAND * MLAND) / 256;
  for (int it = 0; it < 6; ++it) {
    k_pinv_gemm<<<pg, 256, 0, stream>>>(attn2, zin, tb2, 1.f);
    k_diag_sub<<<eb, 256, 0, stream>>>(tb2, tb3, 7.f);
    k_pinv_gemm<<<pg, 256, 0, stream>>>(tb2, tb3, tb4, 1.f);
    k_diag_sub<<<eb, 256, 0, stream>>>(tb4, tb3, 15.f);
    k_pinv_gemm<<<pg, 256, 0, stream>>>(tb2, tb3, tb4, 1.f);
    k_diag_sub<<<eb, 256, 0, stream>>>(tb4, tb3, 13.f);
    k_pinv_gemm<<<pg, 256, 0, stream>>>(zin, tb3, zout, 0.25f);
    float* tmp = zin; zin = zout; zout = tmp;
  }
  k_zT_bf16<<<eb, 256, 0, stream>>>(zin, zTb);

  k_attn3v<<<dim3(MLAND / 64, BH), 128, 0, stream>>>(qlb, kb, vb, a3vT);

  k_conv<<<(BH * SEQ * DHEAD) / 256, 256, 0, stream>>>(vb, conv_w, oh);
  const size_t fused_smem = (size_t)(256 * 256 + 64 * 256 + 8 * 16 * 264) * sizeof(__bf16);
  k_attn_fused<<<dim3(SEQ / 128, BH), 256, fused_smem, stream>>>(qb, klb, zTb, a3vT, oh);

  // output projection + bias + residual x  -> d_out (f32)
  k_gemm_bf16<2, false><<<dim3(ROWS / 128, DIMF / 256, 1), 256, 0, stream>>>(
      oh, woutb, DIMF, DIMF, DIMF, 0, 0, 0, 1.f,
      out, nullptr, DIMF, b_out, x, nullptr, nullptr, nullptr);
}
